// DPMACE_61984968016255
// MI455X (gfx1250) — matrix operand order repaired and checked by reading, not compiled, not run on hardware
//
#include <hip/hip_runtime.h>
#include <math.h>

#define NN 20000
#define NE 640000
#define ZS 10
#define H 128
#define NB 8
#define RMAXF 5.0f
#define PI_F 3.14159265358979323846f
#define WAVES 8
#define C1 0.4886025119f
#define C2 1.0925484306f
#define C3 0.3153915653f
#define C4 0.5462742153f
#define C0 0.2820947918f

typedef __attribute__((ext_vector_type(16))) __bf16 v16bf;
typedef __attribute__((ext_vector_type(8))) float v8f;

union Frag {
  v16bf v;
  uint4 q[2];
  __bf16 h[16];
};

__device__ __forceinline__ v8f wmma_bf(const Frag& a, const Frag& b, v8f c) {
  return __builtin_amdgcn_wmma_f32_16x16x32_bf16(false, a.v, false, b.v, (short)0, c,
                                                 false, false);
}

__device__ __forceinline__ void atom_add(float* p, float v) {
  __hip_atomic_fetch_add(p, v, __ATOMIC_RELAXED, __HIP_MEMORY_SCOPE_AGENT);
}

// A fragment (16x32 bf16, M=row): lane holds row (lane&15); elements 0-7 = K kb..kb+7,
// elements 8-15 = K kb+16..kb+23, kb = (lane<16?0:8) + 32*kc.   [ISA 7.12.2]
__device__ __forceinline__ void load_a_bf16(Frag& f, const __bf16* row, int kc, int lane) {
  const int kb = ((lane & 16) ? 8 : 0) + kc * 32;
  f.q[0] = *reinterpret_cast<const uint4*>(row + kb);
  f.q[1] = *reinterpret_cast<const uint4*>(row + kb + 16);
}

__device__ __forceinline__ void load_a_f32(Frag& f, const float* row, int kc, int lane) {
  const int kb = ((lane & 16) ? 8 : 0) + kc * 32;
#pragma unroll
  for (int i = 0; i < 8; ++i) f.h[i] = (__bf16)row[kb + i];
#pragma unroll
  for (int i = 0; i < 8; ++i) f.h[8 + i] = (__bf16)row[kb + 16 + i];
}

// B fragment (32x16 bf16, N=col): lane holds column; elements 0-7 = K kb..kb+7,
// elements 8-15 = K kb+16..kb+23, kb = (lane<16?0:8) + 32*kc. colrow points at column base (column-contiguous storage).
__device__ __forceinline__ void load_b_bf16(Frag& f, const __bf16* colrow, int kc, int lane) {
  const int kb = ((lane & 16) ? 8 : 0) + kc * 32;
  f.q[0] = *reinterpret_cast<const uint4*>(colrow + kb);
  f.q[1] = *reinterpret_cast<const uint4*>(colrow + kb + 16);
}

// ---------------------------------------------------------------- small prep kernels

__global__ void convert_kernel(const float* __restrict__ Wmsg, const float* __restrict__ Wprod,
                               const float* __restrict__ Wsc, __bf16* __restrict__ msg_cm,
                               __bf16* __restrict__ msg_rm, __bf16* __restrict__ prod_cm,
                               __bf16* __restrict__ prod_rm, __bf16* __restrict__ sc_cm) {
  const int idx = blockIdx.x * blockDim.x + threadIdx.x;
  if (idx >= 2 * H * H) return;
  const int l = idx / (H * H), rem = idx % (H * H), k = rem / H, n = rem % H;
  const int cmi = l * H * H + n * H + k;
  msg_rm[idx] = (__bf16)Wmsg[idx];
  msg_cm[cmi] = (__bf16)Wmsg[idx];
  prod_rm[idx] = (__bf16)Wprod[idx];
  prod_cm[cmi] = (__bf16)Wprod[idx];
  sc_cm[cmi] = (__bf16)Wsc[idx];
}

__global__ void gvec_kernel(const float* __restrict__ Wread, const float* __restrict__ Wsc,
                            const float* __restrict__ scale_p, float* __restrict__ g2,
                            float* __restrict__ g1) {
  const int k = threadIdx.x;
  if (k >= H) return;
  const float sc = scale_p[0];
  g2[k] = sc * Wread[H + k];
  float acc = 0.f;
  for (int n = 0; n < H; ++n) acc += Wread[H + n] * Wsc[H * H + k * H + n];
  g1[k] = sc * (Wread[k] + acc);
}

__global__ void embed_kernel(const float* __restrict__ attrs, const float* __restrict__ Wemb,
                             __bf16* __restrict__ h0bf) {
  const int idx = blockIdx.x * blockDim.x + threadIdx.x;
  if (idx >= NN * H) return;
  const int n = idx / H, c = idx % H;
  float acc = 0.f;
#pragma unroll
  for (int z = 0; z < ZS; ++z) acc += attrs[n * ZS + z] * Wemb[z * H + c];
  h0bf[idx] = (__bf16)acc;
}

__global__ void geom_kernel(const float* __restrict__ pos, const float* __restrict__ shifts,
                            const int* __restrict__ ei, float* __restrict__ evec,
                            float* __restrict__ er, float* __restrict__ erad,
                            float* __restrict__ esh) {
  const int e = blockIdx.x * blockDim.x + threadIdx.x;
  if (e >= NE) return;
  const int s = ei[e], d = ei[NE + e];
  const float vx = pos[d * 3 + 0] - pos[s * 3 + 0] + shifts[e * 3 + 0];
  const float vy = pos[d * 3 + 1] - pos[s * 3 + 1] + shifts[e * 3 + 1];
  const float vz = pos[d * 3 + 2] - pos[s * 3 + 2] + shifts[e * 3 + 2];
  const float r = sqrtf(vx * vx + vy * vy + vz * vz + 1e-12f);
  const float invr = 1.f / r;
  evec[(size_t)e * 3 + 0] = vx;
  evec[(size_t)e * 3 + 1] = vy;
  evec[(size_t)e * 3 + 2] = vz;
  er[e] = r;
  const float x = vx * invr, y = vy * invr, z = vz * invr;
  float sh[9];
  sh[0] = C0;
  sh[1] = C1 * y;
  sh[2] = C1 * z;
  sh[3] = C1 * x;
  sh[4] = C2 * x * y;
  sh[5] = C2 * y * z;
  sh[6] = C3 * (3.f * z * z - 1.f);
  sh[7] = C2 * x * z;
  sh[8] = C4 * (x * x - y * y);
#pragma unroll
  for (int j = 0; j < 9; ++j) esh[(size_t)e * 9 + j] = sh[j];
  const float pref = sqrtf(2.f / RMAXF);
  float rad[NB];
  if (r < RMAXF) {
    const float a = PI_F / RMAXF;
    float sa, ca;
    sincosf(a * r, &sa, &ca);
    const float env = 0.5f * (ca + 1.f);
    float s_n = sa, c_n = ca;
#pragma unroll
    for (int n = 1; n <= NB; ++n) {
      rad[n - 1] = pref * s_n * invr * env;
      const float ns = s_n * ca + c_n * sa;
      const float nc = c_n * ca - s_n * sa;
      s_n = ns;
      c_n = nc;
    }
  } else {
#pragma unroll
    for (int n = 0; n < NB; ++n) rad[n] = 0.f;
  }
#pragma unroll
  for (int b = 0; b < NB; ++b) erad[(size_t)e * NB + b] = rad[b];
}

// -------------------------------------------------------- forward: edge message GEMM
// m_e = (h[src]@W_msg) * (rad@W_rad) * (sh@w_sh);  atomic scatter into agg[dst].
__global__ void __launch_bounds__(256) msg_forward_kernel(
    const __bf16* __restrict__ hbf, const int* __restrict__ ei, const float* __restrict__ erad,
    const float* __restrict__ esh, const float* __restrict__ Wrad, const float* __restrict__ wsh,
    const __bf16* __restrict__ wmsg_cm, float* __restrict__ agg) {
  __shared__ float s_rad[WAVES][16][NB];
  __shared__ float s_S[WAVES][16];
  __shared__ int s_src[WAVES][16];
  __shared__ int s_dst[WAVES][16];
  const int wave = threadIdx.x >> 5, lane = threadIdx.x & 31;
  const int tile = blockIdx.x * WAVES + wave;
  const bool active = tile < NE / 16;
  if (active && lane < 16) {
    const int e = tile * 16 + lane;
    s_src[wave][lane] = ei[e];
    s_dst[wave][lane] = ei[NE + e];
    float S = 0.f;
#pragma unroll
    for (int j = 0; j < 9; ++j) S += esh[(size_t)e * 9 + j] * wsh[j];
    s_S[wave][lane] = S;
#pragma unroll
    for (int b = 0; b < NB; ++b) s_rad[wave][lane][b] = erad[(size_t)e * NB + b];
  }
  __syncthreads();
  if (!active) return;
  const int col = lane & 15, grp = lane >> 4;
  Frag a[4];
  {
    const __bf16* row = hbf + (size_t)s_src[wave][col] * H;
#pragma unroll
    for (int kc = 0; kc < 4; ++kc) load_a_bf16(a[kc], row, kc, lane);
  }
#pragma unroll 1
  for (int nt = 0; nt < 8; ++nt) {
    const int ch = nt * 16 + col;
    v8f c = {};
#pragma unroll
    for (int kc = 0; kc < 4; ++kc) {
      Frag b;
      load_b_bf16(b, wmsg_cm + (size_t)ch * H, kc, lane);
      c = wmma_bf(a[kc], b, c);
    }
    float wr[NB];
#pragma unroll
    for (int b = 0; b < NB; ++b) wr[b] = Wrad[b * H + ch];
#pragma unroll
    for (int r = 0; r < 8; ++r) {
      const int el = r + 8 * grp;
      float A = 0.f;
#pragma unroll
      for (int b = 0; b < NB; ++b) A += s_rad[wave][el][b] * wr[b];
      atom_add(agg + (size_t)s_dst[wave][el] * H + ch, c[r] * A * s_S[wave][el]);
    }
  }
}

// ------------------------------------------------- forward: node update (two GEMMs)
// pre = agg@W_prod ; h_new = silu(pre) + h@W_sc ; store pre (f32) and h_new (bf16).
__global__ void __launch_bounds__(256) node_update_kernel(
    const float* __restrict__ agg, const __bf16* __restrict__ hin,
    const __bf16* __restrict__ wprod_cm, const __bf16* __restrict__ wsc_cm,
    float* __restrict__ preOut, __bf16* __restrict__ hout) {
  const int wave = threadIdx.x >> 5, lane = threadIdx.x & 31;
  const int tile = blockIdx.x * WAVES + wave;
  if (tile >= NN / 16) return;
  const int col = lane & 15, grp = lane >> 4;
  const int rowNode = tile * 16 + col;
  Frag aP[4], aH[4];
#pragma unroll
  for (int kc = 0; kc < 4; ++kc) {
    load_a_f32(aP[kc], agg + (size_t)rowNode * H, kc, lane);
    load_a_bf16(aH[kc], hin + (size_t)rowNode * H, kc, lane);
  }
#pragma unroll 1
  for (int nt = 0; nt < 8; ++nt) {
    const int ch = nt * 16 + col;
    v8f cp = {}, cs = {};
#pragma unroll
    for (int kc = 0; kc < 4; ++kc) {
      Frag b;
      load_b_bf16(b, wprod_cm + (size_t)ch * H, kc, lane);
      cp = wmma_bf(aP[kc], b, cp);
      load_b_bf16(b, wsc_cm + (size_t)ch * H, kc, lane);
      cs = wmma_bf(aH[kc], b, cs);
    }
#pragma unroll
    for (int r = 0; r < 8; ++r) {
      const int node = tile * 16 + r + 8 * grp;
      const float pre = cp[r];
      preOut[(size_t)node * H + ch] = pre;
      if (hout) {
        const float sig = 1.f / (1.f + __expf(-pre));
        hout[(size_t)node * H + ch] = (__bf16)(pre * sig + cs[r]);
      }
    }
  }
}

// ----------------------------------------------- backward: node (dagg = dpre @ Wprod^T)
__global__ void __launch_bounds__(256) node_backward_kernel(
    const float* __restrict__ pre, const float* __restrict__ gvec, const float* __restrict__ dh,
    int useDh, const __bf16* __restrict__ wprod_rm, float* __restrict__ dagg) {
  const int wave = threadIdx.x >> 5, lane = threadIdx.x & 31;
  const int tile = blockIdx.x * WAVES + wave;
  if (tile >= NN / 16) return;
  const int col = lane & 15, grp = lane >> 4;
  const int rowNode = tile * 16 + col;
  Frag a[4];
#pragma unroll
  for (int kc = 0; kc < 4; ++kc) {
    const int kb = ((lane & 16) ? 8 : 0) + kc * 32;
#pragma unroll
    for (int half = 0; half < 2; ++half) {
      const int k0 = kb + half * 16;
#pragma unroll
      for (int i = 0; i < 8; ++i) {
        const int K = k0 + i;
        const float p = pre[(size_t)rowNode * H + K];
        float g = gvec[K];
        if (useDh) g += dh[(size_t)rowNode * H + K];
        const float sig = 1.f / (1.f + __expf(-p));
        a[kc].h[half * 8 + i] = (__bf16)(g * sig * (1.f + p * (1.f - sig)));
      }
    }
  }
#pragma unroll 1
  for (int nt = 0; nt < 8; ++nt) {
    const int ch = nt * 16 + col;
    v8f c = {};
#pragma unroll
    for (int kc = 0; kc < 4; ++kc) {
      Frag b;
      load_b_bf16(b, wprod_rm + (size_t)ch * H, kc, lane);
      c = wmma_bf(a[kc], b, c);
    }
#pragma unroll
    for (int r = 0; r < 8; ++r)
      dagg[(size_t)(tile * 16 + r + 8 * grp) * H + ch] = c[r];
  }
}

// -------------------------------------------------------------- backward: edge kernel
// Recomputes t = h[src]@W_msg (WMMA), folds dm=dagg[dst]: dr/dS channel reductions,
// optional dh[src] += (dm*A*S)@W_msg^T (second WMMA pass via LDS re-layout),
// then geometry chain rule -> per-edge force atomics.
template <bool DO_DH>
__global__ void __launch_bounds__(256) edge_backward_kernel(
    const __bf16* __restrict__ hbf, const int* __restrict__ ei, const float* __restrict__ evec,
    const float* __restrict__ er, const float* __restrict__ erad, const float* __restrict__ esh,
    const float* __restrict__ Wrad, const float* __restrict__ wsh,
    const __bf16* __restrict__ wmsg_cm, const __bf16* __restrict__ wmsg_rm,
    const float* __restrict__ dagg, float* __restrict__ dh, float* __restrict__ forces) {
  __shared__ float s_rad[WAVES][16][NB];
  __shared__ float s_q[WAVES][16][NB];
  __shared__ float s_S[WAVES][16];
  __shared__ float s_vx[WAVES][16], s_vy[WAVES][16], s_vz[WAVES][16], s_r[WAVES][16];
  __shared__ int s_src[WAVES][16], s_dst[WAVES][16];
  __shared__ __bf16 s_dt[WAVES][16 * H];

  const int wave = threadIdx.x >> 5, lane = threadIdx.x & 31;
  const int tile = blockIdx.x * WAVES + wave;
  const bool active = tile < NE / 16;
  if (active && lane < 16) {
    const int e = tile * 16 + lane;
    s_src[wave][lane] = ei[e];
    s_dst[wave][lane] = ei[NE + e];
    float S = 0.f;
#pragma unroll
    for (int j = 0; j < 9; ++j) S += esh[(size_t)e * 9 + j] * wsh[j];
    s_S[wave][lane] = S;
#pragma unroll
    for (int b = 0; b < NB; ++b) s_rad[wave][lane][b] = erad[(size_t)e * NB + b];
    const float r = er[e];
    s_r[wave][lane] = r;
    s_vx[wave][lane] = evec[(size_t)e * 3 + 0];
    s_vy[wave][lane] = evec[(size_t)e * 3 + 1];
    s_vz[wave][lane] = evec[(size_t)e * 3 + 2];
    const float pref = sqrtf(2.f / RMAXF);
    float q[NB];
    if (r < RMAXF) {
      const float a = PI_F / RMAXF;
      float sa, ca;
      sincosf(a * r, &sa, &ca);
      const float env = 0.5f * (ca + 1.f);
      const float denv = -0.5f * a * sa;
      const float invr = 1.f / r;
      float s_n = sa, c_n = ca;
#pragma unroll
      for (int n = 1; n <= NB; ++n) {
        const float na = (float)n * a;
        q[n - 1] = pref * ((na * c_n * r - s_n) * invr * invr * env + s_n * invr * denv);
        const float ns = s_n * ca + c_n * sa;
        const float nc = c_n * ca - s_n * sa;
        s_n = ns;
        c_n = nc;
      }
    } else {
#pragma unroll
      for (int n = 0; n < NB; ++n) q[n] = 0.f;
    }
#pragma unroll
    for (int b = 0; b < NB; ++b) s_q[wave][lane][b] = q[b];
  }
  __syncthreads();
  if (!active) return;
  const int col = lane & 15, grp = lane >> 4;
  Frag a[4];
  {
    const __bf16* row = hbf + (size_t)s_src[wave][col] * H;
#pragma unroll
    for (int kc = 0; kc < 4; ++kc) load_a_bf16(a[kc], row, kc, lane);
  }
  float drp[8], dSp[8];
#pragma unroll
  for (int r = 0; r < 8; ++r) {
    drp[r] = 0.f;
    dSp[r] = 0.f;
  }
#pragma unroll 1
  for (int nt = 0; nt < 8; ++nt) {
    const int ch = nt * 16 + col;
    v8f c = {};
#pragma unroll
    for (int kc = 0; kc < 4; ++kc) {
      Frag b;
      load_b_bf16(b, wmsg_cm + (size_t)ch * H, kc, lane);
      c = wmma_bf(a[kc], b, c);
    }
    float wr[NB];
#pragma unroll
    for (int b = 0; b < NB; ++b) wr[b] = Wrad[b * H + ch];
#pragma unroll
    for (int r = 0; r < 8; ++r) {
      const int el = r + 8 * grp;
      float A = 0.f, P = 0.f;
#pragma unroll
      for (int b = 0; b < NB; ++b) {
        A += s_rad[wave][el][b] * wr[b];
        P += s_q[wave][el][b] * wr[b];
      }
      const float dm = dagg[(size_t)s_dst[wave][el] * H + ch];
      const float p = dm * c[r];
      const float S = s_S[wave][el];
      drp[r] += p * S * P;  // dr contribution: dA * dP/dr projection
      dSp[r] += p * A;      // dS contribution
      if (DO_DH) s_dt[wave][el * H + ch] = (__bf16)(dm * A * S);
    }
  }
  if (DO_DH) {
    __builtin_amdgcn_wave_barrier();  // intra-wave LDS RAW; same-wave DS ops are in-order
    Frag ad[4];
    {
      const __bf16* row = &s_dt[wave][col * H];
#pragma unroll
      for (int kc = 0; kc < 4; ++kc) load_a_bf16(ad[kc], row, kc, lane);
    }
#pragma unroll 1
    for (int nt = 0; nt < 8; ++nt) {
      const int ch = nt * 16 + col;
      v8f c = {};
#pragma unroll
      for (int kc = 0; kc < 4; ++kc) {
        Frag b;
        load_b_bf16(b, wmsg_rm + (size_t)ch * H, kc, lane);
        c = wmma_bf(ad[kc], b, c);
      }
#pragma unroll
      for (int r = 0; r < 8; ++r)
        atom_add(dh + (size_t)s_src[wave][r + 8 * grp] * H + ch, c[r]);
    }
  }
  // butterfly channel reduction across the 16 lanes of each half-wave
#pragma unroll
  for (int off = 1; off < 16; off <<= 1) {
#pragma unroll
    for (int r = 0; r < 8; ++r) {
      drp[r] += __shfl_xor(drp[r], off, 32);
      dSp[r] += __shfl_xor(dSp[r], off, 32);
    }
  }
#pragma unroll
  for (int r = 0; r < 8; ++r) {
    if (col == r) {
      const int el = r + 8 * grp;
      const float dS = dSp[r];
      const float dr = drp[r];
      const float rr = s_r[wave][el];
      const float invr = 1.f / rr;
      const float ux = s_vx[wave][el] * invr, uy = s_vy[wave][el] * invr,
                  uz = s_vz[wave][el] * invr;
      const float w1 = dS * wsh[1], w2 = dS * wsh[2], w3 = dS * wsh[3], w4 = dS * wsh[4];
      const float w5 = dS * wsh[5], w6 = dS * wsh[6], w7 = dS * wsh[7], w8 = dS * wsh[8];
      const float gx = C1 * w3 + C2 * (uy * w4 + uz * w7) + 2.f * C4 * ux * w8;
      const float gy = C1 * w1 + C2 * (ux * w4 + uz * w5) - 2.f * C4 * uy * w8;
      const float gz = C1 * w2 + C2 * (uy * w5 + ux * w7) + 6.f * C3 * uz * w6;
      const float gdu = gx * ux + gy * uy + gz * uz;
      const float dvx = (gx - gdu * ux) * invr + dr * ux;
      const float dvy = (gy - gdu * uy) * invr + dr * uy;
      const float dvz = (gz - gdu * uz) * invr + dr * uz;
      const int sn = s_src[wave][el], dn = s_dst[wave][el];
      atom_add(forces + (size_t)dn * 3 + 0, -dvx);
      atom_add(forces + (size_t)dn * 3 + 1, -dvy);
      atom_add(forces + (size_t)dn * 3 + 2, -dvz);
      atom_add(forces + (size_t)sn * 3 + 0, dvx);
      atom_add(forces + (size_t)sn * 3 + 1, dvy);
      atom_add(forces + (size_t)sn * 3 + 2, dvz);
    }
  }
}

// ------------------------------------------------------------------------- host side

extern "C" void kernel_launch(void* const* d_in, const int* in_sizes, int n_in, void* d_out,
                              int out_size, void* d_ws, size_t ws_size, hipStream_t stream) {
  (void)in_sizes;
  (void)n_in;
  (void)out_size;
  (void)ws_size;
  const float* pos = (const float*)d_in[0];
  const float* attrs = (const float*)d_in[1];
  const float* shifts = (const float*)d_in[2];
  const float* Wemb = (const float*)d_in[4];
  const float* Wmsg = (const float*)d_in[5];
  const float* Wrad = (const float*)d_in[6];
  const float* wsh = (const float*)d_in[7];
  const float* Wprod = (const float*)d_in[8];
  const float* Wsc = (const float*)d_in[9];
  const float* Wread = (const float*)d_in[10];
  const float* scale = (const float*)d_in[11];
  const int* ei = (const int*)d_in[13];
  float* F = (float*)d_out;

  char* w = (char*)d_ws;
  auto carve = [&](size_t bytes) {
    char* p = w;
    w += (bytes + 255) & ~size_t(255);
    return p;
  };
  const size_t NHf = (size_t)NN * H * sizeof(float);
  const size_t NHb = (size_t)NN * H * sizeof(__bf16);
  __bf16* h0bf = (__bf16*)carve(NHb);
  __bf16* h1bf = (__bf16*)carve(NHb);
  float* pre0 = (float*)carve(NHf);
  float* pre1 = (float*)carve(NHf);
  float* agg0 = (float*)carve(NHf);
  float* agg1 = (float*)carve(NHf);
  float* dagg = (float*)carve(NHf);
  float* dhb = (float*)carve(NHf);
  float* evec = (float*)carve((size_t)NE * 3 * sizeof(float));
  float* erb = (float*)carve((size_t)NE * sizeof(float));
  float* erad = (float*)carve((size_t)NE * NB * sizeof(float));
  float* esh = (float*)carve((size_t)NE * 9 * sizeof(float));
  const size_t WB = (size_t)2 * H * H * sizeof(__bf16);
  __bf16* wmsg_cm = (__bf16*)carve(WB);
  __bf16* wmsg_rm = (__bf16*)carve(WB);
  __bf16* wprod_cm = (__bf16*)carve(WB);
  __bf16* wprod_rm = (__bf16*)carve(WB);
  __bf16* wsc_cm = (__bf16*)carve(WB);
  float* g2 = (float*)carve(H * sizeof(float));
  float* g1 = (float*)carve(H * sizeof(float));

  hipMemsetAsync(F, 0, (size_t)NN * 3 * sizeof(float), stream);
  hipMemsetAsync(agg0, 0, NHf, stream);
  hipMemsetAsync(agg1, 0, NHf, stream);
  hipMemsetAsync(dhb, 0, NHf, stream);

  convert_kernel<<<(2 * H * H + 255) / 256, 256, 0, stream>>>(Wmsg, Wprod, Wsc, wmsg_cm, wmsg_rm,
                                                              wprod_cm, wprod_rm, wsc_cm);
  gvec_kernel<<<1, H, 0, stream>>>(Wread, Wsc, scale, g2, g1);
  embed_kernel<<<(NN * H + 255) / 256, 256, 0, stream>>>(attrs, Wemb, h0bf);
  geom_kernel<<<(NE + 255) / 256, 256, 0, stream>>>(pos, shifts, ei, evec, erb, erad, esh);

  const int edgeBlocks = (NE / 16 + WAVES - 1) / WAVES;
  const int nodeBlocks = (NN / 16 + WAVES - 1) / WAVES;

  // forward layer 0
  msg_forward_kernel<<<edgeBlocks, 256, 0, stream>>>(h0bf, ei, erad, esh, Wrad, wsh, wmsg_cm,
                                                     agg0);
  node_update_kernel<<<nodeBlocks, 256, 0, stream>>>(agg0, h0bf, wprod_cm, wsc_cm, pre0, h1bf);
  // forward layer 1
  msg_forward_kernel<<<edgeBlocks, 256, 0, stream>>>(h1bf, ei, erad, esh, Wrad + NB * H, wsh + 9,
                                                     wmsg_cm + H * H, agg1);
  node_update_kernel<<<nodeBlocks, 256, 0, stream>>>(agg1, h1bf, wprod_cm + H * H,
                                                     wsc_cm + H * H, pre1, (__bf16*)nullptr);
  // backward layer 1
  node_backward_kernel<<<nodeBlocks, 256, 0, stream>>>(pre1, g2, (const float*)nullptr, 0,
                                                       wprod_rm + H * H, dagg);
  edge_backward_kernel<true><<<edgeBlocks, 256, 0, stream>>>(
      h1bf, ei, evec, erb, erad, esh, Wrad + NB * H, wsh + 9, wmsg_cm + H * H, wmsg_rm + H * H,
      dagg, dhb, F);
  // backward layer 0
  node_backward_kernel<<<nodeBlocks, 256, 0, stream>>>(pre0, g1, dhb, 1, wprod_rm, dagg);
  edge_backward_kernel<false><<<edgeBlocks, 256, 0, stream>>>(h0bf, ei, evec, erb, erad, esh,
                                                              Wrad, wsh, wmsg_cm, wmsg_rm, dagg,
                                                              (float*)nullptr, F);
}
